// GPTTransformerBlock_37271726195321
// MI455X (gfx1250) — matrix operand order repaired and checked by reading, not compiled, not run on hardware
//
#include <hip/hip_runtime.h>
#include <hip/hip_bf16.h>

// ---------------------------------------------------------------------------
// GPT transformer block for MI455X (gfx1250), bf16 WMMA path.
// B=2, S=2048, E=1024, H=16, D=64, EF=4.
// ---------------------------------------------------------------------------

typedef __bf16 bf16_t;
typedef bf16_t v16bf __attribute__((ext_vector_type(16)));
typedef bf16_t v8bf  __attribute__((ext_vector_type(8)));
typedef bf16_t v4bf  __attribute__((ext_vector_type(4)));
typedef float  v8f   __attribute__((ext_vector_type(8)));

#define B_    2
#define S_    2048
#define E_    1024
#define H_    16
#define D_    64
#define BS_   4096      // B*S
#define NQKV  3072      // 3*H*D
#define NFF   4096      // EF*E
#define NEGI  (-1.0e30f)   // finite "-inf" sentinel: keeps softmax math branchless

// ---------------------------------------------------------------------------
// WMMA helpers (CDNA5 v_wmma_f32_16x16x32_bf16)
// ---------------------------------------------------------------------------
__device__ __forceinline__ v8f wmma_bf16(v16bf a, v16bf b, v8f c) {
    return __builtin_amdgcn_wmma_f32_16x16x32_bf16(
        /*neg_a=*/false, a, /*neg_b=*/false, b,
        /*c_mod=*/(short)0, c, /*reuse_a=*/false, /*reuse_b=*/false);
}

__device__ __forceinline__ v8f vzero8() {
    v8f z;
#pragma unroll
    for (int i = 0; i < 8; i++) z[i] = 0.0f;
    return z;
}

// A-matrix fragment (16x32 bf16, M x K). Per ISA: lane half 0 holds
// K = {0..7, 16..23}, lane half 1 holds K = {8..15, 24..31}; M = lane&15.
__device__ __forceinline__ v16bf fragA(const bf16_t* base, int stride, int row, int lane) {
    const int m = lane & 15, half = lane >> 4;
    const bf16_t* p = base + (size_t)(row + m) * stride;
    v8bf lo = *(const v8bf*)(p + half * 8);
    v8bf hi = *(const v8bf*)(p + 16 + half * 8);
    v16bf f;
#pragma unroll
    for (int i = 0; i < 8; i++) { f[i] = lo[i]; f[8 + i] = hi[i]; }
    return f;
}

// B-matrix fragment (32x16 bf16, K x N), read from B^T storage [N][K].
// Per ISA: lane half 0 holds K = {0..7, 16..23}, lane half 1 holds K = {8..15, 24..31}; N = lane&15.
__device__ __forceinline__ v16bf fragB(const bf16_t* baseT, int stride, int row, int lane) {
    const int n = lane & 15, half = lane >> 4;
    const bf16_t* p = baseT + (size_t)(row + n) * stride + half * 8;
    v8bf lo = *(const v8bf*)(p);
    v8bf hi = *(const v8bf*)(p + 16);
    v16bf f;
#pragma unroll
    for (int i = 0; i < 8; i++) { f[i] = lo[i]; f[8 + i] = hi[i]; }
    return f;
}

// ---------------------------------------------------------------------------
// Tiled transpose + f32->bf16 convert:  Wt[n][k] = W[k][n].
// ---------------------------------------------------------------------------
__global__ void __launch_bounds__(256)
k_transpose_tiled(const float* __restrict__ W, bf16_t* __restrict__ Wt, int K, int N) {
    __shared__ float tile[32][33];
    const int tx = threadIdx.x & 31;
    const int ty = threadIdx.x >> 5;          // 0..7
    const int n0 = blockIdx.x * 32;
    const int k0 = blockIdx.y * 32;
#pragma unroll
    for (int i = 0; i < 4; i++)
        tile[ty + i * 8][tx] = W[(size_t)(k0 + ty + i * 8) * N + n0 + tx];
    __syncthreads();
#pragma unroll
    for (int i = 0; i < 4; i++)
        Wt[(size_t)(n0 + ty + i * 8) * K + k0 + tx] = (bf16_t)tile[tx][ty + i * 8];
}

// Per-head QKV weight transpose: [H,E,D] -> WT[n][e], n = sel*1024 + h*64 + d.
__global__ void __launch_bounds__(256)
k_qkv_transpose_tiled(const float* __restrict__ Wq, const float* __restrict__ Wk,
                      const float* __restrict__ Wv, bf16_t* __restrict__ WT) {
    __shared__ float tile[32][33];
    const int tx = threadIdx.x & 31;
    const int ty = threadIdx.x >> 5;
    const int z = blockIdx.z;                 // sel*16 + h
    const int sel = z >> 4, h = z & 15;
    const float* W = (sel == 0) ? Wq : (sel == 1) ? Wk : Wv;
    const float* src = W + (size_t)h * E_ * D_;   // [E][D]
    const int e0 = blockIdx.x * 32;
    const int d0 = blockIdx.y * 32;
#pragma unroll
    for (int i = 0; i < 4; i++)
        tile[ty + i * 8][tx] = src[(size_t)(e0 + ty + i * 8) * D_ + d0 + tx];
    __syncthreads();
    bf16_t* dst = WT + ((size_t)(sel * 1024 + h * 64 + d0)) * E_;
#pragma unroll
    for (int i = 0; i < 4; i++)
        dst[(size_t)(ty + i * 8) * E_ + e0 + tx] = (bf16_t)tile[tx][ty + i * 8];
}

__global__ void k_bias_concat(const float* __restrict__ bq, const float* __restrict__ bk,
                              const float* __restrict__ bv, float* __restrict__ bcat) {
    const int t = blockIdx.x * 256 + threadIdx.x;
    if (t < NQKV) {
        const int sel = t >> 10, hd = t & 1023;
        const float* bb = (sel == 0) ? bq : (sel == 1) ? bk : bv;
        bcat[t] = bb[hd];
    }
}

// ---------------------------------------------------------------------------
// LayerNorm over E=1024, one row per workgroup, bf16 output.
// ---------------------------------------------------------------------------
__global__ void __launch_bounds__(256)
k_layernorm(const float* __restrict__ X, const float* __restrict__ g,
            const float* __restrict__ b, bf16_t* __restrict__ O) {
    const int row = blockIdx.x;
    const int t = threadIdx.x;
    const float4 v = ((const float4*)(X + (size_t)row * E_))[t];
    __shared__ float rs[256], rq[256];
    rs[t] = v.x + v.y + v.z + v.w;
    rq[t] = v.x * v.x + v.y * v.y + v.z * v.z + v.w * v.w;
    __syncthreads();
    for (int o = 128; o > 0; o >>= 1) {
        if (t < o) { rs[t] += rs[t + o]; rq[t] += rq[t + o]; }
        __syncthreads();
    }
    const float mean = rs[0] * (1.0f / E_);
    const float var  = rq[0] * (1.0f / E_) - mean * mean;
    const float rstd = rsqrtf(var + 1e-5f);
    const float4 gg = ((const float4*)g)[t];
    const float4 bb = ((const float4*)b)[t];
    v4bf o4;
    o4[0] = (bf16_t)((v.x - mean) * rstd * gg.x + bb.x);
    o4[1] = (bf16_t)((v.y - mean) * rstd * gg.y + bb.y);
    o4[2] = (bf16_t)((v.z - mean) * rstd * gg.z + bb.z);
    o4[3] = (bf16_t)((v.w - mean) * rstd * gg.w + bb.w);
    ((v4bf*)(O + (size_t)row * E_))[t] = o4;
}

// ---------------------------------------------------------------------------
// Tiled WMMA GEMM: C[M,N] = act(A[M,K] @ B[K,N] + bias) (+resid).
// Block tile 128x128x32; 8 waves, each 32x64 (2x4 WMMA tiles).
// Double-buffered LDS, one barrier per K-step, prefetch 2 tiles ahead.
// ---------------------------------------------------------------------------
#define LDT 40   // LDS tile stride (32 + 8 pad, keeps 16B alignment)

template <bool RELU, bool RESID, bool OUTF, bool OUTB>
__global__ void __launch_bounds__(256)
k_gemm_bf16(const bf16_t* __restrict__ A, int lda,
            const bf16_t* __restrict__ Bt, int ldb,
            const float* __restrict__ bias,
            const float* __restrict__ resid,
            float* __restrict__ Cf, bf16_t* __restrict__ Cb,
            int M, int N, int K) {
    __shared__ bf16_t As[2][128 * LDT];
    __shared__ bf16_t Bs[2][128 * LDT];

    const int tid  = threadIdx.x;
    const int lane = tid & 31;
    const int wave = tid >> 5;
    const int m0 = blockIdx.y * 128;
    const int n0 = blockIdx.x * 128;
    const int wm = (wave >> 1) * 32;
    const int wn = (wave & 1) * 64;

    v8f acc[2][4];
#pragma unroll
    for (int i = 0; i < 2; i++)
#pragma unroll
        for (int j = 0; j < 4; j++) acc[i][j] = vzero8();

    const int lrow = tid >> 1;           // 0..127
    const int lcol = (tid & 1) * 16;     // 0 or 16
    const bf16_t* ag = A  + (size_t)(m0 + lrow) * lda + lcol;
    const bf16_t* bg = Bt + (size_t)(n0 + lrow) * ldb + lcol;
    const int lofs = lrow * LDT + lcol;

    // Prologue: stage K-tile 0 into buffer 0.
    {
        v8bf a0 = *(const v8bf*)(ag);
        v8bf a1 = *(const v8bf*)(ag + 8);
        v8bf b0 = *(const v8bf*)(bg);
        v8bf b1 = *(const v8bf*)(bg + 8);
        *(v8bf*)&As[0][lofs]     = a0;
        *(v8bf*)&As[0][lofs + 8] = a1;
        *(v8bf*)&Bs[0][lofs]     = b0;
        *(v8bf*)&Bs[0][lofs + 8] = b1;
    }
    __syncthreads();

    int cur = 0;
    for (int k0 = 0; k0 < K; k0 += 32) {
        const bool more = (k0 + 32) < K;
        v8bf na0, na1, nb0, nb1;
        if (more) {
            na0 = *(const v8bf*)(ag + k0 + 32);
            na1 = *(const v8bf*)(ag + k0 + 40);
            nb0 = *(const v8bf*)(bg + k0 + 32);
            nb1 = *(const v8bf*)(bg + k0 + 40);
            if (k0 + 96 <= K) {   // prefetch two tiles ahead (global_prefetch_b8)
                __builtin_prefetch(ag + k0 + 64, 0, 0);
                __builtin_prefetch(bg + k0 + 64, 0, 0);
            }
        }

        v16bf af[2], bfv[4];
#pragma unroll
        for (int i = 0; i < 2; i++) af[i]  = fragA(As[cur], LDT, wm + i * 16, lane);
#pragma unroll
        for (int j = 0; j < 4; j++) bfv[j] = fragB(Bs[cur], LDT, wn + j * 16, lane);
#pragma unroll
        for (int i = 0; i < 2; i++)
#pragma unroll
            for (int j = 0; j < 4; j++)
                acc[i][j] = wmma_bf16(af[i], bfv[j], acc[i][j]);

        if (more) {
            const int nxt = cur ^ 1;
            *(v8bf*)&As[nxt][lofs]     = na0;
            *(v8bf*)&As[nxt][lofs + 8] = na1;
            *(v8bf*)&Bs[nxt][lofs]     = nb0;
            *(v8bf*)&Bs[nxt][lofs + 8] = nb1;
            __syncthreads();
            cur = nxt;
        }
    }

    // Epilogue. C layout: reg r -> M = (lane>>4)*8 + r, N = lane&15.
    const int nlane = lane & 15;
    const int mbase = (lane >> 4) * 8;
#pragma unroll
    for (int i = 0; i < 2; i++) {
        const int mrow = m0 + wm + i * 16 + mbase;
#pragma unroll
        for (int j = 0; j < 4; j++) {
            const int ncol = n0 + wn + j * 16 + nlane;
            const float bsv = bias[ncol];
#pragma unroll
            for (int r = 0; r < 8; r++) {
                float c = acc[i][j][r] + bsv;
                if constexpr (RELU) c = fmaxf(c, 0.0f);
                const size_t off = (size_t)(mrow + r) * N + ncol;
                if constexpr (RESID) c += resid[off];
                if constexpr (OUTF) Cf[off] = c;
                if constexpr (OUTB) Cb[off] = (bf16_t)c;
            }
        }
    }
}

// ---------------------------------------------------------------------------
// Attention. qkv bf16 layout [BS][3072]: q at +0, k at +1024, v at +2048,
// per-head column offset h*64. Reference quirk: mask keeps (s >= t) AND
// (score != 0); softmax over s per row t; o[s] = sum_t p[t][s] * v[t].
// ---------------------------------------------------------------------------

// Pass 1: per-row (max, sumexp) over kept columns. One wave per 16 t-rows.
// Fully branchless online softmax using a finite -1e30 sentinel (no exec-mask
// divergence, no -inf NaN cases).
__global__ void __launch_bounds__(256)
k_attn_stats(const bf16_t* __restrict__ qkv, float2* __restrict__ rst) {
    const int bh = blockIdx.x;           // b*H + h
    const int bb = bh / H_, h = bh % H_;
    const int lane = threadIdx.x & 31;
    const int wave = threadIdx.x >> 5;
    const int t0 = blockIdx.y * 128 + wave * 16;
    const float scale = 0.125f;          // D^-0.5

    const bf16_t* qbase = qkv + ((size_t)(bb * S_ + t0)) * NQKV + h * D_;
    const v16bf aq0 = fragA(qbase,      NQKV, 0, lane);   // d 0..31
    const v16bf aq1 = fragA(qbase + 32, NQKV, 0, lane);   // d 32..63

    const int nlane = lane & 15;
    const int mbase = (lane >> 4) * 8;

    float mx[8], sm[8];
#pragma unroll
    for (int r = 0; r < 8; r++) { mx[r] = NEGI; sm[r] = 0.0f; }

    for (int s0 = t0; s0 < S_; s0 += 16) {
        const bf16_t* kbase = qkv + ((size_t)(bb * S_ + s0)) * NQKV + 1024 + h * D_;
        const v16bf bk0 = fragB(kbase,      NQKV, 0, lane);
        const v16bf bk1 = fragB(kbase + 32, NQKV, 0, lane);
        v8f sc = vzero8();
        sc = wmma_bf16(aq0, bk0, sc);
        sc = wmma_bf16(aq1, bk1, sc);
#pragma unroll
        for (int r = 0; r < 8; r++) {
            const int t = t0 + mbase + r;
            const int s = s0 + nlane;
            const float raw = sc[r];
            const bool keep = (s >= t) && (raw != 0.0f);
            const float v  = keep ? raw * scale : NEGI;
            const float nm = fmaxf(mx[r], v);
            const float e  = keep ? __expf(v - nm) : 0.0f;
            sm[r] = sm[r] * __expf(mx[r] - nm) + e;
            mx[r] = nm;
        }
    }
    // Reduce across the 16-lane group (columns of one row live in 16 lanes).
#pragma unroll
    for (int r = 0; r < 8; r++) {
#pragma unroll
        for (int off = 1; off < 16; off <<= 1) {
            const float om = __shfl_xor(mx[r], off, 16);
            const float os = __shfl_xor(sm[r], off, 16);
            const float nm = fmaxf(mx[r], om);
            sm[r] = sm[r] * __expf(mx[r] - nm) + os * __expf(om - nm);
            mx[r] = nm;
        }
    }
    if (nlane == 0) {
#pragma unroll
        for (int r = 0; r < 8; r++)
            rst[(size_t)bh * S_ + t0 + mbase + r] = make_float2(mx[r], sm[r]);
    }
}

// Pass 2: o[s][d] = sum_{t<=s} p[t][s] * v[t][d], accumulated with WMMA by
// transposing p through LDS (bf16) and staging v transposed as B^T[d][t].
__global__ void __launch_bounds__(256)
k_attn_out(const bf16_t* __restrict__ qkv, const float2* __restrict__ rst,
           bf16_t* __restrict__ obf) {
    __shared__ bf16_t pT[128 * LDT];   // [s_local][t_local]
    __shared__ bf16_t vT[64 * LDT];    // [d][t_local]
    __shared__ float2 rstl[32];        // (rmax, 1/rsum) for current t-block

    const int bh = blockIdx.x;
    const int bb = bh / H_, h = bh % H_;
    const int sblk = blockIdx.y * 128;
    const int tid  = threadIdx.x;
    const int lane = tid & 31;
    const int wave = tid >> 5;
    const int nlane = lane & 15;
    const int mbase = (lane >> 4) * 8;
    const float scale = 0.125f;

    const int sw = sblk + wave * 16;   // this wave's 16 output rows (s)

    // k fragments for this wave's fixed s-columns (loaded once).
    const bf16_t* kbase = qkv + ((size_t)(bb * S_ + sw)) * NQKV + 1024 + h * D_;
    const v16bf bk0 = fragB(kbase,      NQKV, 0, lane);
    const v16bf bk1 = fragB(kbase + 32, NQKV, 0, lane);

    v8f acc[4];
#pragma unroll
    for (int j = 0; j < 4; j++) acc[j] = vzero8();

    // vT staging mapping: one v8bf global load + 8 transposed ds stores.
    const int vt_t    = tid & 31;      // t within block
    const int vt_doct = tid >> 5;      // d octet 0..7

    const int tend = sblk + 128;
    for (int tb = 0; tb < tend; tb += 32) {
        if (tid < 32) {
            const float2 st = rst[(size_t)bh * S_ + tb + tid];
            rstl[tid] = make_float2(st.x, (st.y > 0.0f) ? 1.0f / st.y : 0.0f);
        }
        {
            const v8bf vv = *(const v8bf*)(qkv +
                ((size_t)(bb * S_ + tb + vt_t)) * NQKV + 2048 + h * D_ + vt_doct * 8);
#pragma unroll
            for (int i = 0; i < 8; i++)
                vT[(vt_doct * 8 + i) * LDT + vt_t] = vv[i];
        }
        __syncthreads();

        // p for 32 t-rows x this wave's 16 s-columns; store transposed.
        // A lane's 8 elements are 8 consecutive t-columns -> one b128 store.
#pragma unroll
        for (int tt = 0; tt < 2; tt++) {
            const bf16_t* qbase =
                qkv + ((size_t)(bb * S_ + tb + tt * 16)) * NQKV + h * D_;
            const v16bf aq0 = fragA(qbase,      NQKV, 0, lane);
            const v16bf aq1 = fragA(qbase + 32, NQKV, 0, lane);
            v8f sc = vzero8();
            sc = wmma_bf16(aq0, bk0, sc);
            sc = wmma_bf16(aq1, bk1, sc);
            v8bf pk;
#pragma unroll
            for (int r = 0; r < 8; r++) {
                const int tl = tt * 16 + mbase + r;   // 0..31
                const int t  = tb + tl;
                const int s  = sw + nlane;
                const float raw = sc[r];
                const bool keep = (s >= t) && (raw != 0.0f);
                const float2 st = rstl[tl];
                const float p = keep ? __expf(raw * scale - st.x) * st.y : 0.0f;
                pk[r] = (bf16_t)p;
            }
            *(v8bf*)&pT[(wave * 16 + nlane) * LDT + tt * 16 + mbase] = pk;
        }
        __syncthreads();

        // o += p^T @ v : A = pT rows (s), K = 32 (t); B^T = vT rows (d).
        const v16bf ap = fragA(pT, LDT, wave * 16, lane);
#pragma unroll
        for (int j = 0; j < 4; j++) {
            const v16bf bv = fragB(vT, LDT, j * 16, lane);
            acc[j] = wmma_bf16(ap, bv, acc[j]);
        }
        __syncthreads();
    }

    // Store o as bf16 [BS][E] with column h*64 + d (heads concatenated).
#pragma unroll
    for (int j = 0; j < 4; j++) {
#pragma unroll
        for (int r = 0; r < 8; r++) {
            const int s = sw + mbase + r;
            const int d = j * 16 + nlane;
            obf[((size_t)(bb * S_ + s)) * E_ + h * D_ + d] = (bf16_t)acc[j][r];
        }
    }
}

// ---------------------------------------------------------------------------
// Host-side orchestration (graph-capture safe: stream launches only).
// ---------------------------------------------------------------------------
extern "C" void kernel_launch(void* const* d_in, const int* in_sizes, int n_in,
                              void* d_out, int out_size, void* d_ws, size_t ws_size,
                              hipStream_t stream) {
    (void)in_sizes; (void)n_in; (void)out_size; (void)ws_size;
    const float* x   = (const float*)d_in[0];
    const float* Wq  = (const float*)d_in[1];
    const float* bq  = (const float*)d_in[2];
    const float* Wk  = (const float*)d_in[3];
    const float* bk  = (const float*)d_in[4];
    const float* Wv  = (const float*)d_in[5];
    const float* bv  = (const float*)d_in[6];
    const float* Wp  = (const float*)d_in[7];
    const float* bp  = (const float*)d_in[8];
    const float* W1  = (const float*)d_in[9];
    const float* b1  = (const float*)d_in[10];
    const float* W2  = (const float*)d_in[11];
    const float* b2  = (const float*)d_in[12];
    const float* g1  = (const float*)d_in[13];
    const float* be1 = (const float*)d_in[14];
    const float* g2  = (const float*)d_in[15];
    const float* be2 = (const float*)d_in[16];

    char* ws = (char*)d_ws;
    size_t off = 0;
    auto alloc = [&](size_t bytes) -> void* {
        void* p = ws + off;
        off = (off + bytes + 255) & ~(size_t)255;
        return p;
    };
    bf16_t* wqkvT = (bf16_t*)alloc((size_t)NQKV * E_ * sizeof(bf16_t));
    bf16_t* wpT   = (bf16_t*)alloc((size_t)E_ * E_ * sizeof(bf16_t));
    bf16_t* w1T   = (bf16_t*)alloc((size_t)NFF * E_ * sizeof(bf16_t));
    bf16_t* w2T   = (bf16_t*)alloc((size_t)E_ * NFF * sizeof(bf16_t));
    float*  bqkv  = (float*)alloc((size_t)NQKV * sizeof(float));
    bf16_t* hbf   = (bf16_t*)alloc((size_t)BS_ * E_ * sizeof(bf16_t));
    bf16_t* qkvbf = (bf16_t*)alloc((size_t)BS_ * NQKV * sizeof(bf16_t));
    float2* rstat = (float2*)alloc((size_t)B_ * H_ * S_ * sizeof(float2));
    bf16_t* obf   = (bf16_t*)alloc((size_t)BS_ * E_ * sizeof(bf16_t));
    float*  x1    = (float*)alloc((size_t)BS_ * E_ * sizeof(float));
    bf16_t* h2bf  = (bf16_t*)alloc((size_t)BS_ * E_ * sizeof(bf16_t));
    bf16_t* a1bf  = (bf16_t*)alloc((size_t)BS_ * NFF * sizeof(bf16_t));

    // 0) Weight conversion/transposition (bf16, [N][K])
    k_qkv_transpose_tiled<<<dim3(E_ / 32, D_ / 32, 48), 256, 0, stream>>>(Wq, Wk, Wv, wqkvT);
    k_bias_concat<<<NQKV / 256, 256, 0, stream>>>(bq, bk, bv, bqkv);
    k_transpose_tiled<<<dim3(E_ / 32,  E_ / 32),  256, 0, stream>>>(Wp, wpT, E_, E_);
    k_transpose_tiled<<<dim3(NFF / 32, E_ / 32),  256, 0, stream>>>(W1, w1T, E_, NFF);
    k_transpose_tiled<<<dim3(E_ / 32,  NFF / 32), 256, 0, stream>>>(W2, w2T, NFF, E_);

    // 1) LN1
    k_layernorm<<<BS_, 256, 0, stream>>>(x, g1, be1, hbf);

    // 2) Fused QKV projection: [4096,3072] = hbf @ Wqkv + bqkv
    k_gemm_bf16<false, false, false, true>
        <<<dim3(NQKV / 128, BS_ / 128), 256, 0, stream>>>(
            hbf, E_, wqkvT, E_, bqkv, nullptr, nullptr, qkvbf, BS_, NQKV, E_);

    // 3) Attention (row stats, then p^T @ v)
    k_attn_stats<<<dim3(B_ * H_, S_ / 128), 256, 0, stream>>>(qkvbf, rstat);
    k_attn_out<<<dim3(B_ * H_, S_ / 128), 256, 0, stream>>>(qkvbf, rstat, obf);

    // 4) Output projection + residual: x1 = x + o @ Wp + bp
    k_gemm_bf16<false, true, true, false>
        <<<dim3(E_ / 128, BS_ / 128), 256, 0, stream>>>(
            obf, E_, wpT, E_, bp, x, x1, nullptr, BS_, E_, E_);

    // 5) LN2
    k_layernorm<<<BS_, 256, 0, stream>>>(x1, g2, be2, h2bf);

    // 6) MLP up + ReLU
    k_gemm_bf16<true, false, false, true>
        <<<dim3(NFF / 128, BS_ / 128), 256, 0, stream>>>(
            h2bf, E_, w1T, E_, b1, nullptr, nullptr, a1bf, BS_, NFF, E_);

    // 7) MLP down + residual -> d_out (f32)
    k_gemm_bf16<false, true, true, false>
        <<<dim3(E_ / 128, BS_ / 128), 256, 0, stream>>>(
            a1bf, NFF, w2T, NFF, b2, x1, (float*)d_out, nullptr, BS_, E_, NFF);
}
